// MultiHeadAttention_55834574848033
// MI455X (gfx1250) — matrix operand order repaired and checked by reading, not compiled, not run on hardware
//
#include <hip/hip_runtime.h>

// ---------------------------------------------------------------------------
// MHA forward for MI455X (gfx1250, wave32, WMMA + async-LDS double buffering
// + TDM). fp32 inputs -> bf16 operands, fp32 accum via v_wmma_f32_16x16x32_bf16.
// ---------------------------------------------------------------------------

typedef unsigned short u16;
typedef __attribute__((ext_vector_type(16))) __bf16 v16bf;
typedef __attribute__((ext_vector_type(8)))  float  v8f;
typedef __attribute__((ext_vector_type(8)))  unsigned short v8u16;
typedef __attribute__((ext_vector_type(4)))  unsigned int v4u;
typedef __attribute__((ext_vector_type(4)))  int v4i;
typedef __attribute__((ext_vector_type(8)))  int v8i;

#define B_  2
#define T_  2048
#define E_  1024
#define H_  8
#define D_  512
#define HD_ 4096        // H_*D_
#define M_  4096        // B_*T_

#if defined(__has_builtin)
#if __has_builtin(__builtin_amdgcn_tensor_load_to_lds) && \
    __has_builtin(__builtin_amdgcn_s_wait_tensorcnt)
#define HAVE_TDM 1
#endif
#endif
#ifndef HAVE_TDM
#define HAVE_TDM 0
#endif

// ---- bf16 helpers (round-to-nearest-even) ---------------------------------
__device__ __forceinline__ u16 f2bf(float f) {
  union { float f; unsigned u; } x; x.f = f;
  unsigned r = x.u + 0x7FFFu + ((x.u >> 16) & 1u);
  return (u16)(r >> 16);
}
__device__ __forceinline__ float bf2f(u16 b) {
  union { unsigned u; float f; } x; x.u = ((unsigned)b) << 16; return x.f;
}

// ---- CDNA5 async copy: global -> LDS, 16B per lane, ASYNCcnt tracked ------
__device__ __forceinline__ void async_b128(const void* gsrc, void* ldst) {
  unsigned lds = (unsigned)(size_t)ldst;                 // LDS aperture: addr[31:0]
  unsigned long long ga = (unsigned long long)gsrc;
  asm volatile("global_load_async_to_lds_b128 %0, %1, off"
               :: "v"(lds), "v"(ga) : "memory");
}
__device__ __forceinline__ void async_b128_off16(const void* gsrc, void* ldst) {
  // INST_OFFSET is added to BOTH the LDS and global addresses (ISA 08 §4.4)
  unsigned lds = (unsigned)(size_t)ldst;
  unsigned long long ga = (unsigned long long)gsrc;
  asm volatile("global_load_async_to_lds_b128 %0, %1, off offset:16"
               :: "v"(lds), "v"(ga) : "memory");
}

// ---- WMMA fragment loaders (contiguous k order, identical for A and B) ------
// A (16x32, MxK): lanes 0-15 hold row M=lane, K=0..15;
//                 lanes 16-31 hold row M=lane-16, K=16..31.
__device__ __forceinline__ v16bf frag_a(const u16* p0, int ld, int lane) {
  const int l  = lane & 15;
  const int hi = (lane >> 4) & 1;
  const u16* p = p0 + (size_t)l * ld;
  union { v16bf bf; v8u16 h[2]; } f;
  f.h[0] = *(const v8u16*)(p + hi * 16);
  f.h[1] = *(const v8u16*)(p + hi * 16 + 8);
  return f.bf;
}
// B (32x16, KxN) from an N-major source (row n holds the contraction values):
// lanes 0-15 hold column N=lane, K=0..15; lanes 16-31 hold N=lane-16, K=16..31.
__device__ __forceinline__ v16bf frag_b(const u16* p0, int ld, int lane) {
  const int l  = lane & 15;
  const int kh = (lane >> 4) & 1;
  const u16* p = p0 + (size_t)l * ld + kh * 16;
  union { v16bf bf; v8u16 h[2]; } f;
  f.h[0] = *(const v8u16*)(p);
  f.h[1] = *(const v8u16*)(p + 8);
  return f.bf;
}

// ---------------------------------------------------------------------------
// fp32 -> bf16 conversion (vectorized)
// ---------------------------------------------------------------------------
__global__ __launch_bounds__(256) void cvt_kernel(const float* __restrict__ in,
                                                  u16* __restrict__ out, int n) {
  int i = (blockIdx.x * blockDim.x + threadIdx.x) * 4;
  if (i >= n) return;
  float4 v = *(const float4*)(in + i);
  out[i + 0] = f2bf(v.x);
  out[i + 1] = f2bf(v.y);
  out[i + 2] = f2bf(v.z);
  out[i + 3] = f2bf(v.w);
}

// ---------------------------------------------------------------------------
// bf16 WMMA GEMM: C[M,N] = A[M,K] @ Bw[N,K]^T   (both row-major, f32 accum)
// 128x128 block tile, 8 waves (4x2), 32x64 wave tile, K step 32.
// Double-buffered LDS staging with global_load_async_to_lds_b128:
// the DMA for tile n+1 runs concurrently with the WMMAs on tile n, joined by
// a partial s_wait_asynccnt (ASYNCcnt decrements in issue order, ISA 01 §7.1).
// ---------------------------------------------------------------------------
__global__ __launch_bounds__(256) void gemm_bf16_kernel(
    const u16* __restrict__ A, const u16* __restrict__ Bw,
    float* __restrict__ Cf, u16* __restrict__ Cb,
    int Mtot, int Ntot, int Ktot) {
  __shared__ __align__(16) u16 As[2][128 * 32];
  __shared__ __align__(16) u16 Bs[2][128 * 32];

  const int tid  = threadIdx.x;
  const int wave = tid >> 5;
  const int lane = tid & 31;
  const int l    = lane & 15;
  const int hi   = lane >> 4;
  const int wm   = wave >> 1;       // 0..3
  const int wn   = wave & 1;        // 0..1
  const int bm   = blockIdx.y * 128;
  const int bn   = blockIdx.x * 128;

  const int lr = tid >> 1;          // 0..127 (row staged by this thread)
  const int lc = (tid & 1) * 16;    // 0 or 16

  const u16* asrc0 = A  + (size_t)(bm + lr) * Ktot + lc;
  const u16* bsrc0 = Bw + (size_t)(bn + lr) * Ktot + lc;

  // Issue the 4 async b128 copies (32B of A row + 32B of B row) for one K step.
  auto stage = [&](int buf, int k) {
    const u16* as = asrc0 + k;
    const u16* bs = bsrc0 + k;
    async_b128      (as, &As[buf][lr * 32 + lc]);
    async_b128_off16(as, &As[buf][lr * 32 + lc]);
    async_b128      (bs, &Bs[buf][lr * 32 + lc]);
    async_b128_off16(bs, &Bs[buf][lr * 32 + lc]);
  };

  v8f acc[2][4] = {};

  stage(0, 0);                        // prologue: fill buffer 0
  const int nsteps = Ktot / 32;
  for (int s = 0; s < nsteps; ++s) {
    const int cur  = s & 1;
    const bool more = (s + 1) < nsteps;
    if (more) {
      stage(cur ^ 1, (s + 1) * 32);   // DMA next tile while we compute
      __builtin_prefetch(asrc0 + (s + 2) * 32, 0, 1);   // warm L2 one more ahead
      __builtin_prefetch(bsrc0 + (s + 2) * 32, 0, 1);
      asm volatile("s_wait_asynccnt 0x4" ::: "memory"); // current tile landed
    } else {
      asm volatile("s_wait_asynccnt 0x0" ::: "memory");
    }
    __syncthreads();                  // every wave's share of 'cur' is in LDS

#pragma unroll
    for (int mt = 0; mt < 2; ++mt) {
      v16bf a = frag_a(As[cur] + (wm * 32 + mt * 16) * 32, 32, lane);
#pragma unroll
      for (int nt = 0; nt < 4; ++nt) {
        v16bf bb = frag_b(Bs[cur] + (wn * 64 + nt * 16) * 32, 32, lane);
        acc[mt][nt] = __builtin_amdgcn_wmma_f32_16x16x32_bf16(
            false, a, false, bb, (short)0, acc[mt][nt], false, false);
      }
    }
    __syncthreads();                  // done reading 'cur' before it is re-staged
  }

#pragma unroll
  for (int mt = 0; mt < 2; ++mt)
#pragma unroll
    for (int nt = 0; nt < 4; ++nt)
#pragma unroll
      for (int r = 0; r < 8; ++r) {
        int row = bm + wm * 32 + mt * 16 + hi * 8 + r;
        int col = bn + wn * 64 + nt * 16 + l;
        float v = acc[mt][nt][r];
        if (Cf) Cf[(size_t)row * Ntot + col] = v;
        else    Cb[(size_t)row * Ntot + col] = f2bf(v);
      }
}

// ---------------------------------------------------------------------------
// RoPE (half-split) applied in place on bf16 Q/K laid out [B*T, H*D].
// ---------------------------------------------------------------------------
__global__ __launch_bounds__(256) void rope_kernel(u16* __restrict__ X) {
  int i = blockIdx.x * blockDim.x + threadIdx.x;   // over B*T*H*(D/2)
  if (i >= B_ * T_ * H_ * (D_ / 2)) return;
  int j  = i & 255;                 // frequency index 0..255
  int h  = (i >> 8) & (H_ - 1);
  int bt = i >> 11;                 // b*T + t
  int t  = bt & (T_ - 1);
  size_t base = (size_t)bt * HD_ + h * D_ + j;
  float x1 = bf2f(X[base]);
  float x2 = bf2f(X[base + D_ / 2]);
  // inv_freq = ROPE_BASE^(-j/256) ; ln(10000)/256 = 0.0359778918...
  float ang = (float)t * __expf(-0.0359778918f * (float)j);
  float s, c;
  __sincosf(ang, &s, &c);
  X[base]          = f2bf(x1 * c - x2 * s);
  X[base + D_ / 2] = f2bf(x1 * s + x2 * c);
}

// ---------------------------------------------------------------------------
// Flash-style causal attention, all-WMMA.
// Block = (b, h, 16-row q tile), 8 waves; wave w owns d-slice [w*64, w*64+64).
// Q tile staged by the Tensor Data Mover (TENSORcnt) when available.
// S = Q K^T contracted over full D=512 (each wave does a 64-wide partial,
// reduced deterministically through LDS), online softmax, O += P V.
// ---------------------------------------------------------------------------
__global__ __launch_bounds__(256) void flash_attn_kernel(
    const u16* __restrict__ Q, const u16* __restrict__ K,
    const u16* __restrict__ V, u16* __restrict__ O) {
  extern __shared__ char smem[];
  u16*   Qs = (u16*)smem;                                   // [16][512]   16384 B
  float* Ss = (float*)(smem + 16384);                       // [16][64]     4096 B
  u16*   Ps = (u16*)(smem + 16384 + 4096);                  // [16][72]     2304 B
  float* aS = (float*)(smem + 16384 + 4096 + 2304);         // [16]           64 B
  float* lS = aS + 16;                                      // [16]           64 B
  float* Sp = lS + 16;                                      // [8][16][64] 32768 B
  u16*   Vt = (u16*)(smem + 16384 + 4096 + 2304 + 128 + 32768); // [8][64][72] 73728 B

  const int tid  = threadIdx.x;
  const int wave = tid >> 5;
  const int lane = tid & 31;
  const int l    = lane & 15;
  const int hi   = lane >> 4;
  const int qt   = blockIdx.x;
  const int h    = blockIdx.y;
  const int b    = blockIdx.z;
  const int qbase = qt * 16;
  const float scale = 0.04419417382f;   // 1/sqrt(512)

  // ---- Stage the 16x512 bf16 Q tile into LDS -------------------------------
#if HAVE_TDM
  if (wave == 0) {
    // 2D tensor DMA descriptor (D#): data_size=2B, tile 512x16, row stride 4096.
    const u16* qsrc = Q + (size_t)(b * T_ + qbase) * HD_ + h * D_;
    unsigned long long ga = (unsigned long long)qsrc;
    v4u g0;
    g0.x = 1u;                                           // count=1, user mode
    g0.y = (unsigned)(size_t)Qs;                         // lds_addr (bytes)
    g0.z = (unsigned)(ga & 0xFFFFFFFFu);                 // global_addr[31:0]
    g0.w = (unsigned)((ga >> 32) & 0x01FFFFFFu) | 0x80000000u; // [56:32] | type=2
    v8i g1;
    g1[0] = (int)(1u << 16);          // workgroup_mask=0, data_size=1 (2 bytes)
    g1[1] = (int)(512u << 16);        // tensor_dim0[15:0]=512 (in [63:48])
    g1[2] = (int)(16u << 16);         // tensor_dim0[31:16]=0 | tensor_dim1=16
    g1[3] = (int)(512u << 16);        // tensor_dim1[31:16]=0 | tile_dim0=512
    g1[4] = 16;                       // tile_dim1=16, tile_dim2=0
    g1[5] = 4096;                     // tensor_dim0_stride[31:0] = HD_
    g1[6] = 0;                        // stride hi, tensor_dim1_stride lo
    g1[7] = 0;
    v4i gz = {0, 0, 0, 0};
#if __clang_major__ >= 23
    v8i z8 = {};
    __builtin_amdgcn_tensor_load_to_lds(g0, g1, gz, gz, z8, 0);
#else
    __builtin_amdgcn_tensor_load_to_lds(g0, g1, gz, gz, 0);
#endif
    __builtin_amdgcn_s_wait_tensorcnt(0);
  }
#else
  {
    int r = tid >> 4;            // 0..15
    int c = (tid & 15) * 32;     // 0..480
    const u16* src = Q + (size_t)(b * T_ + qbase + r) * HD_ + h * D_ + c;
    u16* dst = Qs + r * 512 + c;
#pragma unroll
    for (int i = 0; i < 4; ++i)
      *(v8u16*)(dst + i * 8) = *(const v8u16*)(src + i * 8);
  }
#endif

  v8f o0 = {}, o1 = {}, o2 = {}, o3 = {};     // 16 rows x 64 d-cols per wave
  float m_row = -1e30f, l_row = 0.f;           // stats for row (lane&15)

  const int nkv = (qbase >> 6) + 1;            // causal: only kv tiles <= q tile
  for (int it = 0; it < nkv; ++it) {
    const int kv0 = it * 64;
    __syncthreads();   // protects Qs (first iter) and Vt/Ps from prior iter

    // Stage this wave's transposed V slice: Vt[d - w*64][kv] (ld = 72).
    {
      u16* VtW = Vt + wave * 64 * 72;
#pragma unroll
      for (int rr = 0; rr < 2; ++rr) {
        int kvr = lane * 2 + rr;
        const u16* src = V + (size_t)(b * T_ + kv0 + kvr) * HD_ + h * D_ + wave * 64;
#pragma unroll
        for (int ch = 0; ch < 8; ++ch) {
          v8u16 u = *(const v8u16*)(src + ch * 8);
#pragma unroll
          for (int e = 0; e < 8; ++e)
            VtW[(ch * 8 + e) * 72 + kvr] = u[e];
        }
      }
    }

    // Warm L2 with the next KV tile's K rows for this wave's d-slice.
    if (it + 1 < nkv) {
      const u16* knext = K + (size_t)(b * T_ + kv0 + 64 + (lane & 15)) * HD_
                           + h * D_ + wave * 64;
      __builtin_prefetch(knext, 0, 1);
    }

    // Partial S over this wave's d-slice: 4 col tiles x 2 K-steps = 8 WMMAs.
    v8f s0 = {}, s1 = {}, s2 = {}, s3 = {};
    const int d0 = wave * 64;
#pragma unroll
    for (int ks = 0; ks < 2; ++ks) {
      v16bf a = frag_a(Qs + d0 + ks * 32, 512, lane);
      const u16* kb = K + (size_t)(b * T_ + kv0) * HD_ + h * D_ + d0 + ks * 32;
      v16bf b0 = frag_b(kb + (size_t)0 * 16 * HD_, HD_, lane);
      s0 = __builtin_amdgcn_wmma_f32_16x16x32_bf16(false, a, false, b0, (short)0, s0, false, false);
      v16bf b1 = frag_b(kb + (size_t)1 * 16 * HD_, HD_, lane);
      s1 = __builtin_amdgcn_wmma_f32_16x16x32_bf16(false, a, false, b1, (short)0, s1, false, false);
      v16bf b2 = frag_b(kb + (size_t)2 * 16 * HD_, HD_, lane);
      s2 = __builtin_amdgcn_wmma_f32_16x16x32_bf16(false, a, false, b2, (short)0, s2, false, false);
      v16bf b3 = frag_b(kb + (size_t)3 * 16 * HD_, HD_, lane);
      s3 = __builtin_amdgcn_wmma_f32_16x16x32_bf16(false, a, false, b3, (short)0, s3, false, false);
    }

    // Write per-wave partials (C layout: row = r + 8*hi, col = c*16 + l).
    {
      float* SpW = Sp + wave * 16 * 64;
#pragma unroll
      for (int r = 0; r < 8; ++r) {
        int row = hi * 8 + r;
        SpW[row * 64 + 0 * 16 + l] = s0[r];
        SpW[row * 64 + 1 * 16 + l] = s1[r];
        SpW[row * 64 + 2 * 16 + l] = s2[r];
        SpW[row * 64 + 3 * 16 + l] = s3[r];
      }
    }
    __syncthreads();

    // Deterministic cross-wave reduction of the 16x64 score tile.
    for (int i = tid; i < 16 * 64; i += 256) {
      float s = 0.f;
#pragma unroll
      for (int w = 0; w < 8; ++w) s += Sp[w * 1024 + i];
      Ss[i] = s;
    }
    __syncthreads();

    // Online softmax stats (computed redundantly by every wave; identical).
    {
      float vals[32];
      float mloc = -1e30f;
      const int grow = qbase + l;                // this lane's query position
#pragma unroll
      for (int c = 0; c < 32; ++c) {
        int col = hi * 32 + c;
        float v = Ss[l * 64 + col] * scale;
        if (kv0 + col > grow) v = -1e30f;        // causal mask
        vals[c] = v;
        mloc = fmaxf(mloc, v);
      }
      mloc = fmaxf(mloc, __shfl_xor(mloc, 16, 32));   // join the two halves
      float m_new = fmaxf(m_row, mloc);
      float alpha = __expf(m_row - m_new);
      float psum = 0.f;
#pragma unroll
      for (int c = 0; c < 32; ++c) {
        float p = __expf(vals[c] - m_new);
        psum += p;
        if (wave == 0) Ps[l * 72 + hi * 32 + c] = f2bf(p);
      }
      psum += __shfl_xor(psum, 16, 32);
      l_row = l_row * alpha + psum;
      m_row = m_new;
      if (wave == 0 && hi == 0) aS[l] = alpha;
    }
    __syncthreads();

    // Rescale running O by alpha(row), then O += P @ V (8 WMMAs per wave).
#pragma unroll
    for (int r = 0; r < 8; ++r) {
      float al = aS[hi * 8 + r];
      o0[r] *= al; o1[r] *= al; o2[r] *= al; o3[r] *= al;
    }
    const u16* VtW = Vt + wave * 64 * 72;
#pragma unroll
    for (int ks = 0; ks < 2; ++ks) {
      v16bf a = frag_a(Ps + ks * 32, 72, lane);
      v16bf v0 = frag_b(VtW + (0 * 16) * 72 + ks * 32, 72, lane);
      o0 = __builtin_amdgcn_wmma_f32_16x16x32_bf16(false, a, false, v0, (short)0, o0, false, false);
      v16bf v1 = frag_b(VtW + (1 * 16) * 72 + ks * 32, 72, lane);
      o1 = __builtin_amdgcn_wmma_f32_16x16x32_bf16(false, a, false, v1, (short)0, o1, false, false);
      v16bf v2 = frag_b(VtW + (2 * 16) * 72 + ks * 32, 72, lane);
      o2 = __builtin_amdgcn_wmma_f32_16x16x32_bf16(false, a, false, v2, (short)0, o2, false, false);
      v16bf v3 = frag_b(VtW + (3 * 16) * 72 + ks * 32, 72, lane);
      o3 = __builtin_amdgcn_wmma_f32_16x16x32_bf16(false, a, false, v3, (short)0, o3, false, false);
    }
  }

  // Epilogue: O /= l(row), store bf16 attention output [B*T, H*D].
  if (wave == 0 && hi == 0) lS[l] = l_row;
  __syncthreads();
#pragma unroll
  for (int r = 0; r < 8; ++r) {
    int row = hi * 8 + r;
    float inv = 1.f / lS[row];
    size_t base = (size_t)(b * T_ + qbase + row) * HD_ + h * D_ + wave * 64;
    O[base + 0 * 16 + l] = f2bf(o0[r] * inv);
    O[base + 1 * 16 + l] = f2bf(o1[r] * inv);
    O[base + 2 * 16 + l] = f2bf(o2[r] * inv);
    O[base + 3 * 16 + l] = f2bf(o3[r] * inv);
  }
}

// ---------------------------------------------------------------------------
// Host orchestration
// ---------------------------------------------------------------------------
extern "C" void kernel_launch(void* const* d_in, const int* in_sizes, int n_in,
                              void* d_out, int out_size, void* d_ws, size_t ws_size,
                              hipStream_t stream) {
  (void)in_sizes; (void)n_in; (void)out_size; (void)ws_size;
  const float* x  = (const float*)d_in[0];
  const float* Wq = (const float*)d_in[1];
  const float* Wk = (const float*)d_in[2];
  const float* Wv = (const float*)d_in[3];
  const float* Wo = (const float*)d_in[4];
  float* out = (float*)d_out;

  char* ws = (char*)d_ws;
  size_t off = 0;
  auto take = [&](size_t bytes) -> char* {
    char* p = ws + off;
    off += (bytes + 255) & ~(size_t)255;
    return p;
  };
  u16* xb  = (u16*)take((size_t)M_ * E_ * 2);    //   8 MB
  u16* Wqb = (u16*)take((size_t)HD_ * E_ * 2);   //   8 MB
  u16* Wkb = (u16*)take((size_t)HD_ * E_ * 2);
  u16* Wvb = (u16*)take((size_t)HD_ * E_ * 2);
  u16* Wob = (u16*)take((size_t)E_ * HD_ * 2);
  u16* Qb  = (u16*)take((size_t)M_ * HD_ * 2);   //  32 MB each
  u16* Kb  = (u16*)take((size_t)M_ * HD_ * 2);
  u16* Vb  = (u16*)take((size_t)M_ * HD_ * 2);
  u16* Ab  = (u16*)take((size_t)M_ * HD_ * 2);

  const int nx = M_ * E_;
  const int nw = HD_ * E_;
  cvt_kernel<<<nx / 4 / 256, 256, 0, stream>>>(x,  xb,  nx);
  cvt_kernel<<<nw / 4 / 256, 256, 0, stream>>>(Wq, Wqb, nw);
  cvt_kernel<<<nw / 4 / 256, 256, 0, stream>>>(Wk, Wkb, nw);
  cvt_kernel<<<nw / 4 / 256, 256, 0, stream>>>(Wv, Wvb, nw);
  cvt_kernel<<<nw / 4 / 256, 256, 0, stream>>>(Wo, Wob, nw);

  // QKV projections: [4096,1024] @ [4096,1024]^T -> bf16 [4096,4096]
  dim3 g1(HD_ / 128, M_ / 128);
  gemm_bf16_kernel<<<g1, 256, 0, stream>>>(xb, Wqb, nullptr, Qb, M_, HD_, E_);
  gemm_bf16_kernel<<<g1, 256, 0, stream>>>(xb, Wkb, nullptr, Kb, M_, HD_, E_);
  gemm_bf16_kernel<<<g1, 256, 0, stream>>>(xb, Wvb, nullptr, Vb, M_, HD_, E_);

  // RoPE on Q and K (in place; both are fully rewritten by the GEMMs above).
  const int nr = B_ * T_ * H_ * (D_ / 2);
  rope_kernel<<<nr / 256, 256, 0, stream>>>(Qb);
  rope_kernel<<<nr / 256, 256, 0, stream>>>(Kb);

  // Flash attention -> bf16 attention output [B*T, H*D].
  dim3 gf(T_ / 16, H_, B_);
  const size_t smem = 16384 + 4096 + 2304 + 128 + 32768 + 73728;  // 129408 B
  flash_attn_kernel<<<gf, 256, smem, stream>>>(Qb, Kb, Vb, Ab);

  // Output projection: [4096,4096] @ [1024,4096]^T -> f32 d_out [4096,1024]
  dim3 g2(E_ / 128, M_ / 128);
  gemm_bf16_kernel<<<g2, 256, 0, stream>>>(Ab, Wob, out, nullptr, M_, E_, HD_);
}
